// SSM_43636867727681
// MI455X (gfx1250) — matrix operand order repaired and checked by reading, not compiled, not run on hardware
//
#include <hip/hip_runtime.h>

// ---------------------------------------------------------------------------
// Types / WMMA helpers (gfx1250, wave32)
// ---------------------------------------------------------------------------
typedef __attribute__((ext_vector_type(16))) _Float16 v16h;
typedef __attribute__((ext_vector_type(8)))  float    v8f;
typedef __attribute__((ext_vector_type(4)))  float    f4;
typedef __attribute__((ext_vector_type(8)))  _Float16 h8;

// A fragment (16x32 f16), row-major source.
// ISA layout: lanes 0-15 hold M=lane, K halves {kb+0..7, kb+16..23};
//             lanes 16-31 hold M=lane-16, K halves {kb+8..15, kb+24..31}.
__device__ __forceinline__ v16h frag_a(const _Float16* row, int kbase, int hi) {
  v16h a;
  const _Float16* p = row + kbase + hi * 8;
  ((f4*)&a)[0] = *(const f4*)(p);
  ((f4*)&a)[1] = *(const f4*)(p + 16);
  return a;
}

// B fragment (32x16 f16) from weights stored [n][k] (k contiguous, row len = krow).
// Lanes 0-15: N=lane, K {kb+0..7, kb+16..23} ; lanes 16-31: N=lane-16, K {kb+8..15, kb+24..31}.
__device__ __forceinline__ v16h frag_b(const _Float16* w, int krow, int n, int kbase, int hi) {
  v16h b;
  const _Float16* p = w + (size_t)n * krow + kbase + hi * 8;
  ((f4*)&b)[0] = *(const f4*)(p);
  ((f4*)&b)[1] = *(const f4*)(p + 16);
  return b;
}

__device__ __forceinline__ v8f wmma16(v16h a, v16h b, v8f c) {
  return __builtin_amdgcn_wmma_f32_16x16x32_f16(false, a, false, b, (short)0, c, false, false);
}

__device__ __forceinline__ v8f zero8() {
  v8f z = {0.f, 0.f, 0.f, 0.f, 0.f, 0.f, 0.f, 0.f};
  return z;
}

// ---------------------------------------------------------------------------
// Kernel 1: weight conversion / reordering + BN folding
// ---------------------------------------------------------------------------
__global__ __launch_bounds__(256) void prep_kernel(
    const float* __restrict__ w_red, const float* __restrict__ w_c,
    const float* __restrict__ b_c,  const float* __restrict__ bn_g,
    const float* __restrict__ bn_b, const float* __restrict__ bn_m,
    const float* __restrict__ bn_v, const float* __restrict__ w_out,
    const float* __restrict__ b_out, const float* __restrict__ bno_g,
    const float* __restrict__ bno_b, const float* __restrict__ bno_m,
    const float* __restrict__ bno_v, const float* __restrict__ w_f1,
    const float* __restrict__ w_f2,
    _Float16* __restrict__ wred16, _Float16* __restrict__ wc16,
    _Float16* __restrict__ wout16, _Float16* __restrict__ wf1h,
    _Float16* __restrict__ wf2h, float* __restrict__ scale_c,
    float* __restrict__ bias_c, float* __restrict__ scale_o,
    float* __restrict__ bias_o) {
  int tid = blockIdx.x * blockDim.x + threadIdx.x;
  int stride = gridDim.x * blockDim.x;
  for (int i = tid; i < 256 * 64; i += stride) {
    int c = i >> 6, m = i & 63;
    wred16[i] = (_Float16)w_red[m * 256 + c];
  }
  for (int i = tid; i < 3 * 9 * 64 * 64; i += stride) {
    int ci = i & 63, co = (i >> 6) & 63;
    int ls = i >> 12;             // l*9 + s
    int l = ls / 9, s = ls % 9;
    int du = s / 3, dv = s % 3;
    wc16[i] = (_Float16)w_c[(((l * 64 + co) * 64 + ci) * 3 + du) * 3 + dv];
  }
  for (int i = tid; i < 256 * 64; i += stride) wout16[i] = (_Float16)w_out[i];
  for (int i = tid; i < 256 * 256; i += stride) wf1h[i] = (_Float16)w_f1[i];
  for (int i = tid; i < 256 * 256; i += stride) wf2h[i] = (_Float16)w_f2[i];
  for (int i = tid; i < 3 * 64; i += stride) {
    float s = bn_g[i] * rsqrtf(bn_v[i] + 1e-5f);
    scale_c[i] = s;
    bias_c[i] = (b_c[i] - bn_m[i]) * s + bn_b[i];
  }
  for (int i = tid; i < 256; i += stride) {
    float s = bno_g[i] * rsqrtf(bno_v[i] + 1e-5f);
    scale_o[i] = s;
    bias_o[i] = (b_out[i] - bno_m[i]) * s + bno_b[i];
  }
}

// ---------------------------------------------------------------------------
// Kernel 2: q = L2normalize(conv1x1(x, w_red)) -> qbuf f16, layout [b][pos][64]
// ---------------------------------------------------------------------------
__global__ __launch_bounds__(64) void qnorm_kernel(const float* __restrict__ x,
                                                   const _Float16* __restrict__ wred16,
                                                   _Float16* __restrict__ qbuf) {
  __shared__ _Float16 wl[256 * 64];
  int tid = threadIdx.x;
  for (int i = tid; i < 2048; i += 64) ((f4*)wl)[i] = ((const f4*)wred16)[i];
  __syncthreads();

  int b = blockIdx.x >> 6;                       // 64 blocks per image
  int pos = ((blockIdx.x & 63) << 6) + tid;      // 0..4095
  const float* xp = x + (size_t)b * 256 * 4096 + pos;

  float acc[64];
#pragma unroll
  for (int m = 0; m < 64; ++m) acc[m] = 0.f;
  for (int c = 0; c < 256; ++c) {
    float xv = xp[(size_t)c * 4096];
    const _Float16* wr = wl + c * 64;
#pragma unroll
    for (int mv = 0; mv < 8; ++mv) {
      h8 w8 = ((const h8*)wr)[mv];
#pragma unroll
      for (int j = 0; j < 8; ++j) acc[mv * 8 + j] = fmaf(xv, (float)w8[j], acc[mv * 8 + j]);
    }
  }
  float nrm = 0.f;
#pragma unroll
  for (int m = 0; m < 64; ++m) nrm += acc[m] * acc[m];
  float inv = 1.f / fmaxf(sqrtf(nrm), 1e-12f);
  _Float16 outv[64];
#pragma unroll
  for (int m = 0; m < 64; ++m) outv[m] = (_Float16)(acc[m] * inv);
  f4* dst = (f4*)(qbuf + ((size_t)b * 4096 + pos) * 64);
#pragma unroll
  for (int i = 0; i < 8; ++i) dst[i] = ((f4*)outv)[i];
}

// ---------------------------------------------------------------------------
// Kernel 3: self-correlation + 3-layer 3x3 conv stack, all WMMA.
// One block = 16 spatial positions (4x4 tile) of one image, 8 wave32s.
// B fragments (9 shifts x 2 ksteps) live in registers across all M-tiles;
// A addresses are base + constant-immediate DS offsets.
// ---------------------------------------------------------------------------
#define CORR_HALVES (6400 + 50176 + 36864 + 25600 + 9216)
#define CORR_SMEM   (CORR_HALVES * 2)

__global__ __launch_bounds__(256) void corr_kernel(const _Float16* __restrict__ qbuf,
                                                   const _Float16* __restrict__ wc16,
                                                   const float* __restrict__ scale_c,
                                                   const float* __restrict__ bias_c,
                                                   _Float16* __restrict__ ybuf) {
  extern __shared__ _Float16 sm[];
  _Float16* qs   = sm;                 // 10*10*64
  _Float16* ssb  = qs + 6400;         // 16*49*64  [p][u][v][c]
  _Float16* wlds = ssb + 50176;       // 9*64*64   [s][cout][cin]
  _Float16* l1o  = wlds + 36864;      // 16*25*64  [p][u'][v'][c]
  _Float16* l2o  = l1o + 25600;       // 16*9*64   [p][u''][v''][c]

  int tid = threadIdx.x;
  int b = blockIdx.x >> 8;
  int tile = blockIdx.x & 255;
  int py0 = (tile >> 4) << 2;
  int px0 = (tile & 15) << 2;

  // ---- stage q 10x10 patch (zero-padded) ----
  for (int site = tid; site < 100; site += 256) {
    int r = site / 10, c = site % 10;
    int gy = py0 - 3 + r, gx = px0 - 3 + c;
    f4* d = (f4*)(qs + site * 64);
    if (gy >= 0 && gy < 64 && gx >= 0 && gx < 64) {
      const f4* s = (const f4*)(qbuf + ((size_t)b * 4096 + gy * 64 + gx) * 64);
#pragma unroll
      for (int i = 0; i < 8; ++i) d[i] = s[i];
    } else {
      f4 z = {0.f, 0.f, 0.f, 0.f};
#pragma unroll
      for (int i = 0; i < 8; ++i) d[i] = z;
    }
  }
  __syncthreads();

  // ---- self-correlation: ss[p][u][v][c] = relu(q[pos+off][c] * q[pos][c]) ----
  for (int site = tid; site < 784; site += 256) {
    int p = site / 49, uv = site % 49;
    int u = uv / 7, v = uv % 7;
    int pr = p >> 2, pc = p & 3;
    const _Float16* qa = qs + ((pr + u) * 10 + (pc + v)) * 64;
    const _Float16* qc = qs + ((pr + 3) * 10 + (pc + 3)) * 64;
    _Float16* d = ssb + site * 64;
#pragma unroll
    for (int j = 0; j < 8; ++j) {
      h8 a8 = ((const h8*)qa)[j];
      h8 c8 = ((const h8*)qc)[j];
      h8 o;
#pragma unroll
      for (int e = 0; e < 8; ++e) {
        float vv = (float)a8[e] * (float)c8[e];
        o[e] = (_Float16)fmaxf(vv, 0.f);
      }
      ((h8*)d)[j] = o;
    }
  }

  auto load_w = [&](int layer) {
    const f4* src = (const f4*)(wc16 + layer * 36864);
    f4* dst = (f4*)wlds;
    for (int i = tid; i < 4608; i += 256) dst[i] = src[i];
  };

  load_w(0);
  __syncthreads();

  int wave = tid >> 5, lane = tid & 31;
  int hi = lane >> 4, ln = lane & 15;
  int ntile = wave & 3, mg = wave >> 2;
  int n = ntile * 16 + ln;   // output channel owned by this lane

  v16h bf[18];

  // ---- layer 1: M=400 (=16 pos x 25 uv), K=64 per shift, 9 shifts ----
  {
#pragma unroll
    for (int s = 0; s < 9; ++s) {
      bf[2 * s]     = frag_b(wlds + s * 4096, 64, n, 0, hi);
      bf[2 * s + 1] = frag_b(wlds + s * 4096, 64, n, 32, hi);
    }
    float sc = scale_c[n], bs = bias_c[n];
    for (int mt = mg; mt < 25; mt += 2) {
      int mrow = mt * 16 + ln;
      int p = mrow / 25, rem = mrow % 25;
      int up = rem / 5, vp = rem % 5;
      const _Float16* base = ssb + ((p * 7 + up) * 7 + vp) * 64;
      v8f acc = zero8();
#pragma unroll
      for (int s = 0; s < 9; ++s) {
        int du = s / 3, dv = s % 3;
        const _Float16* arow = base + (du * 7 + dv) * 64;   // immediate DS offsets
        acc = wmma16(frag_a(arow, 0, hi), bf[2 * s], acc);
        acc = wmma16(frag_a(arow, 32, hi), bf[2 * s + 1], acc);
      }
#pragma unroll
      for (int r = 0; r < 8; ++r) {
        int me = mt * 16 + r + hi * 8;
        int pp = me / 25, rr = me % 25;
        int uu = rr / 5, vv = rr % 5;
        float val = fmaxf(acc[r] * sc + bs, 0.f);
        l1o[((pp * 5 + uu) * 5 + vv) * 64 + n] = (_Float16)val;
      }
    }
  }
  __syncthreads();
  load_w(1);
  __syncthreads();

  // ---- layer 2: M=144 (=16 pos x 9 uv) ----
  {
#pragma unroll
    for (int s = 0; s < 9; ++s) {
      bf[2 * s]     = frag_b(wlds + s * 4096, 64, n, 0, hi);
      bf[2 * s + 1] = frag_b(wlds + s * 4096, 64, n, 32, hi);
    }
    float sc = scale_c[64 + n], bs = bias_c[64 + n];
    for (int mt = mg; mt < 9; mt += 2) {
      int mrow = mt * 16 + ln;
      int p = mrow / 9, rem = mrow % 9;
      int up = rem / 3, vp = rem % 3;
      const _Float16* base = l1o + ((p * 5 + up) * 5 + vp) * 64;
      v8f acc = zero8();
#pragma unroll
      for (int s = 0; s < 9; ++s) {
        int du = s / 3, dv = s % 3;
        const _Float16* arow = base + (du * 5 + dv) * 64;
        acc = wmma16(frag_a(arow, 0, hi), bf[2 * s], acc);
        acc = wmma16(frag_a(arow, 32, hi), bf[2 * s + 1], acc);
      }
#pragma unroll
      for (int r = 0; r < 8; ++r) {
        int me = mt * 16 + r + hi * 8;
        int pp = me / 9, rr = me % 9;
        int uu = rr / 3, vv = rr % 3;
        float val = fmaxf(acc[r] * sc + bs, 0.f);
        l2o[((pp * 3 + uu) * 3 + vv) * 64 + n] = (_Float16)val;
      }
    }
  }
  __syncthreads();
  load_w(2);
  __syncthreads();

  // ---- layer 3: M=16 (one row per position); mean over 1x1 is identity ----
  if (mg == 0) {
#pragma unroll
    for (int s = 0; s < 9; ++s) {
      bf[2 * s]     = frag_b(wlds + s * 4096, 64, n, 0, hi);
      bf[2 * s + 1] = frag_b(wlds + s * 4096, 64, n, 32, hi);
    }
    float sc = scale_c[128 + n], bs = bias_c[128 + n];
    int p = ln;
    const _Float16* base = l2o + p * 576;   // ((p*3+du)*3+dv)*64 = p*576 + s*64
    v8f acc = zero8();
#pragma unroll
    for (int s = 0; s < 9; ++s) {
      const _Float16* arow = base + s * 64;
      acc = wmma16(frag_a(arow, 0, hi), bf[2 * s], acc);
      acc = wmma16(frag_a(arow, 32, hi), bf[2 * s + 1], acc);
    }
#pragma unroll
    for (int r = 0; r < 8; ++r) {
      int pp = r + hi * 8;
      int gpos = (py0 + (pp >> 2)) * 64 + px0 + (pp & 3);
      float val = fmaxf(acc[r] * sc + bs, 0.f);
      ybuf[((size_t)b * 4096 + gpos) * 64 + n] = (_Float16)val;
    }
  }
}

// ---------------------------------------------------------------------------
// Kernel 4: feat = BN(wout @ y); out1 = x + feat; out = wf2 @ relu(wf1 @ out1)
// One block = 32 positions, 8 wave32s, each wave owns 2 N-tiles of 16 channels.
// ---------------------------------------------------------------------------
__global__ __launch_bounds__(256) void tail_kernel(const float* __restrict__ x,
                                                   const _Float16* __restrict__ ybuf,
                                                   const _Float16* __restrict__ wout16,
                                                   const float* __restrict__ scale_o,
                                                   const float* __restrict__ bias_o,
                                                   const _Float16* __restrict__ wf1,
                                                   const float* __restrict__ bf1,
                                                   const _Float16* __restrict__ wf2,
                                                   const float* __restrict__ bf2,
                                                   float* __restrict__ out) {
  __shared__ _Float16 yl[32 * 64];
  __shared__ _Float16 o1[32 * 256];
  __shared__ _Float16 hb[32 * 256];

  int tid = threadIdx.x;
  int b = blockIdx.x >> 7;                 // 128 blocks per image
  int pos0 = (blockIdx.x & 127) << 5;

  {
    const f4* s = (const f4*)(ybuf + ((size_t)b * 4096 + pos0) * 64);
    f4* d = (f4*)yl;
    for (int i = tid; i < 256; i += 256) d[i] = s[i];
  }
  __syncthreads();

  int wave = tid >> 5, lane = tid & 31;
  int hi = lane >> 4, ln = lane & 15;

  // ---- GEMM1: feat (K=64) + BN fold + residual -> o1 f16 [pos][256] ----
#pragma unroll
  for (int nt = 0; nt < 2; ++nt) {
    int n = (wave * 2 + nt) * 16 + ln;
    v8f a0 = zero8(), a1 = zero8();
#pragma unroll
    for (int ks = 0; ks < 64; ks += 32) {
      v16h bfr = frag_b(wout16, 64, n, ks, hi);
      a0 = wmma16(frag_a(yl + ln * 64, ks, hi), bfr, a0);
      a1 = wmma16(frag_a(yl + (16 + ln) * 64, ks, hi), bfr, a1);
    }
    float sc = scale_o[n], bs = bias_o[n];
    const float* xcol = x + ((size_t)b * 256 + n) * 4096 + pos0;
#pragma unroll
    for (int r = 0; r < 8; ++r) {
      int m0 = r + hi * 8, m1 = 16 + m0;
      o1[m0 * 256 + n] = (_Float16)(a0[r] * sc + bs + xcol[m0]);
      o1[m1 * 256 + n] = (_Float16)(a1[r] * sc + bs + xcol[m1]);
    }
  }
  __syncthreads();

  // ---- GEMM2: h = relu(o1 @ wf1^T + b1), K=256 ----
#pragma unroll
  for (int nt = 0; nt < 2; ++nt) {
    int n = (wave * 2 + nt) * 16 + ln;
    v8f a0 = zero8(), a1 = zero8();
    for (int ks = 0; ks < 256; ks += 32) {
      v16h bfr = frag_b(wf1, 256, n, ks, hi);
      a0 = wmma16(frag_a(o1 + ln * 256, ks, hi), bfr, a0);
      a1 = wmma16(frag_a(o1 + (16 + ln) * 256, ks, hi), bfr, a1);
    }
    float bs = bf1[n];
#pragma unroll
    for (int r = 0; r < 8; ++r) {
      int m0 = r + hi * 8, m1 = 16 + m0;
      hb[m0 * 256 + n] = (_Float16)fmaxf(a0[r] + bs, 0.f);
      hb[m1 * 256 + n] = (_Float16)fmaxf(a1[r] + bs, 0.f);
    }
  }
  __syncthreads();

  // ---- GEMM3: out = h @ wf2^T + b2 (f32 store, NCHW) ----
#pragma unroll
  for (int nt = 0; nt < 2; ++nt) {
    int n = (wave * 2 + nt) * 16 + ln;
    v8f a0 = zero8(), a1 = zero8();
    for (int ks = 0; ks < 256; ks += 32) {
      v16h bfr = frag_b(wf2, 256, n, ks, hi);
      a0 = wmma16(frag_a(hb + ln * 256, ks, hi), bfr, a0);
      a1 = wmma16(frag_a(hb + (16 + ln) * 256, ks, hi), bfr, a1);
    }
    float bs = bf2[n];
    float* ocol = out + ((size_t)b * 256 + n) * 4096 + pos0;
#pragma unroll
    for (int r = 0; r < 8; ++r) {
      int m0 = r + hi * 8, m1 = 16 + m0;
      ocol[m0] = a0[r] + bs;
      ocol[m1] = a1[r] + bs;
    }
  }
}

// ---------------------------------------------------------------------------
extern "C" void kernel_launch(void* const* d_in, const int* in_sizes, int n_in,
                              void* d_out, int out_size, void* d_ws, size_t ws_size,
                              hipStream_t stream) {
  (void)in_sizes; (void)n_in; (void)out_size; (void)ws_size;
  const float* x     = (const float*)d_in[0];
  const float* w_red = (const float*)d_in[1];
  const float* w_c   = (const float*)d_in[2];
  const float* b_c   = (const float*)d_in[3];
  const float* bn_g  = (const float*)d_in[4];
  const float* bn_b  = (const float*)d_in[5];
  const float* bn_m  = (const float*)d_in[6];
  const float* bn_v  = (const float*)d_in[7];
  const float* w_out = (const float*)d_in[8];
  const float* b_out = (const float*)d_in[9];
  const float* bno_g = (const float*)d_in[10];
  const float* bno_b = (const float*)d_in[11];
  const float* bno_m = (const float*)d_in[12];
  const float* bno_v = (const float*)d_in[13];
  const float* w_f1  = (const float*)d_in[14];
  const float* b_f1  = (const float*)d_in[15];
  const float* w_f2  = (const float*)d_in[16];
  const float* b_f2  = (const float*)d_in[17];

  char* w = (char*)d_ws;
  _Float16* qbuf   = (_Float16*)(w + 0);         // 8*4096*64 f16
  _Float16* ybuf   = (_Float16*)(w + 4194304);   // 8*4096*64 f16
  _Float16* wred16 = (_Float16*)(w + 8388608);   // 256*64
  _Float16* wc16   = (_Float16*)(w + 8421376);   // 3*9*64*64
  _Float16* wout16 = (_Float16*)(w + 8642560);   // 256*64
  _Float16* wf1h   = (_Float16*)(w + 8675328);   // 256*256
  _Float16* wf2h   = (_Float16*)(w + 8806400);   // 256*256
  float* scale_c   = (float*)(w + 8937472);      // 3*64
  float* bias_c    = (float*)(w + 8938240);      // 3*64
  float* scale_o   = (float*)(w + 8939008);      // 256
  float* bias_o    = (float*)(w + 8940032);      // 256

  hipFuncSetAttribute(reinterpret_cast<const void*>(&corr_kernel),
                      hipFuncAttributeMaxDynamicSharedMemorySize, CORR_SMEM);

  prep_kernel<<<64, 256, 0, stream>>>(w_red, w_c, b_c, bn_g, bn_b, bn_m, bn_v,
                                      w_out, b_out, bno_g, bno_b, bno_m, bno_v,
                                      w_f1, w_f2, wred16, wc16, wout16, wf1h,
                                      wf2h, scale_c, bias_c, scale_o, bias_o);
  qnorm_kernel<<<512, 64, 0, stream>>>(x, wred16, qbuf);
  corr_kernel<<<2048, 256, CORR_SMEM, stream>>>(qbuf, wc16, scale_c, bias_c, ybuf);
  tail_kernel<<<1024, 256, 0, stream>>>(x, ybuf, wout16, scale_o, bias_o,
                                        wf1h, b_f1, wf2h, b_f2, (float*)d_out);
}
